// AdapterModule_25159918420443
// MI455X (gfx1250) — compile-verified, matrix operand order checked by reading, not run on hardware
//
#include <hip/hip_runtime.h>
#include <hip/hip_bf16.h>

typedef __attribute__((ext_vector_type(16))) _Float16 v16h;
typedef __attribute__((ext_vector_type(8)))  float    v8f;
typedef __attribute__((ext_vector_type(4)))  unsigned int v4u;

union Frag16 { v16h h; v4u u[2]; };

// Load a 16-half WMMA A/B fragment: halves [0..7] and [16..23] relative to p,
// i.e. two 128-bit loads (works for both global and LDS generic pointers).
__device__ __forceinline__ v16h ldfrag(const _Float16* p) {
  Frag16 f;
  f.u[0] = *reinterpret_cast<const v4u*>(p);
  f.u[1] = *reinterpret_cast<const v4u*>(p + 16);
  return f.h;
}

__device__ __forceinline__ v8f wmma_f16(v16h a, v16h b, v8f c) {
  return __builtin_amdgcn_wmma_f32_16x16x32_f16(false, a, false, b, (short)0, c,
                                                false, false);
}

#define NB   1024   // vertices per batch
#define HC   128
#define HID  256
#define CIN  256
#define HW   65536  // 256*256

// ---------------------------------------------------------------- kernel 0
// Convert Wc (128x256), W1 (256x128), W2 (128x256) fp32 -> f16 into workspace.
__global__ __launch_bounds__(256) void k_convert(const float* __restrict__ Wc,
                                                 const float* __restrict__ W1,
                                                 const float* __restrict__ W2,
                                                 _Float16* __restrict__ dst) {
  int i = blockIdx.x * 256 + threadIdx.x;
  if (i < 32768)      dst[i] = (_Float16)Wc[i];
  else if (i < 65536) dst[i] = (_Float16)W1[i - 32768];
  else if (i < 98304) dst[i] = (_Float16)W2[i - 65536];
}

// ---------------------------------------------------------------- kernel 1
// One workgroup (4 waves) per vertex: ROI-align gather -> f16 A in LDS ->
// WMMA GEMM (Wc[128x256] x A[256x64]) -> bias+relu+mean -> pooled f16 [128].
__global__ __launch_bounds__(128) void k_roi_comp(
    const float* __restrict__ feat, const float* __restrict__ verts,
    const _Float16* __restrict__ Wch, const float* __restrict__ bc,
    _Float16* __restrict__ pooled) {
  __shared__ __align__(16) _Float16 Alds[64 * 256];  // [p_pad=64][c=256]

  const int tid = threadIdx.x;
  const int bn  = blockIdx.x;      // b*NB + n
  const int b   = bn >> 10;

  // sampling_ratio=1, unit bins, aligned=True -> single fractional offset
  const float vx = verts[bn * 2 + 0];
  const float vy = verts[bn * 2 + 1];
  const float x0f = vx - 3.5f;
  const float y0f = vy - 3.5f;
  const float xbf = floorf(x0f), ybf = floorf(y0f);
  const float fx = x0f - xbf, fy = y0f - ybf;
  const float gx = 1.f - fx, gy = 1.f - fy;
  const int xbi = (int)xbf, ybi = (int)ybf;

  int cof[8];
#pragma unroll
  for (int j = 0; j < 8; ++j) {
    int x = xbi + j;
    cof[j] = x < 0 ? 0 : (x > 255 ? 255 : x);
  }

  // ---- phase 1: gather 8x8 patch per channel, 2x2 bilinear -> Alds[p][c]
#pragma unroll
  for (int pass = 0; pass < 2; ++pass) {
    const int c = tid + pass * 128;
    const float* fb = feat + ((size_t)(b * CIN + c) << 16);
    float rr[8], pm[7];
    {
      int y = ybi;
      y = y < 0 ? 0 : (y > 255 ? 255 : y);
      const float* rp = fb + (y << 8);
#pragma unroll
      for (int j = 0; j < 8; ++j) rr[j] = rp[cof[j]];
#pragma unroll
      for (int j = 0; j < 7; ++j) pm[j] = gx * rr[j] + fx * rr[j + 1];
    }
#pragma unroll
    for (int i = 1; i < 8; ++i) {
      int y = ybi + i;
      y = y < 0 ? 0 : (y > 255 ? 255 : y);
      const float* rp = fb + (y << 8);
#pragma unroll
      for (int j = 0; j < 8; ++j) rr[j] = rp[cof[j]];
      float cm[7];
#pragma unroll
      for (int j = 0; j < 7; ++j) cm[j] = gx * rr[j] + fx * rr[j + 1];
      const int prow = (i - 1) * 7;
#pragma unroll
      for (int j = 0; j < 7; ++j)
        Alds[(prow + j) * 256 + c] = (_Float16)(gy * pm[j] + fy * cm[j]);
#pragma unroll
      for (int j = 0; j < 7; ++j) pm[j] = cm[j];
    }
#pragma unroll
    for (int p = 49; p < 64; ++p) Alds[p * 256 + c] = (_Float16)0.f;
  }
  __syncthreads();

  // ---- phase 2: comp = Wc x A ; M=128 (2 Mtiles/wave), N=64 (4 Ptiles), K=256
  const int wave  = tid >> 5;
  const int lane  = tid & 31;
  const int lo    = lane & 15;
  const int hiofs = (lane & 16) ? 8 : 0;
  const int hi8   = (lane & 16) ? 8 : 0;

  const v8f vz = {0.f, 0.f, 0.f, 0.f, 0.f, 0.f, 0.f, 0.f};
  v8f acc[2][4];
#pragma unroll
  for (int mt = 0; mt < 2; ++mt)
#pragma unroll
    for (int pt = 0; pt < 4; ++pt) acc[mt][pt] = vz;

  const int mrow0 = (wave * 2 + 0) * 16 + lo;
  const int mrow1 = (wave * 2 + 1) * 16 + lo;
#pragma unroll
  for (int k = 0; k < 8; ++k) {
    const int kb = k * 32 + hiofs;
    v16h a0 = ldfrag(Wch + mrow0 * 256 + kb);
    v16h a1 = ldfrag(Wch + mrow1 * 256 + kb);
#pragma unroll
    for (int pt = 0; pt < 4; ++pt) {
      v16h bf = ldfrag(&Alds[(pt * 16 + lo) * 256 + kb]);
      acc[0][pt] = wmma_f16(a0, bf, acc[0][pt]);
      acc[1][pt] = wmma_f16(a1, bf, acc[1][pt]);
    }
  }

  // ---- phase 3: pooled[o] = mean_{p<49} relu(comp[o,p] + bc[o])
#pragma unroll
  for (int mt = 0; mt < 2; ++mt) {
    const int obase = (wave * 2 + mt) * 16 + hi8;  // D: M = v + 8*hi
    float bcv[8], s[8];
#pragma unroll
    for (int v = 0; v < 8; ++v) { bcv[v] = bc[obase + v]; s[v] = 0.f; }
#pragma unroll
    for (int pt = 0; pt < 4; ++pt) {
      const int p = pt * 16 + lo;  // D: N = lane%16
      const bool ok = p < 49;
#pragma unroll
      for (int v = 0; v < 8; ++v) {
        float t = fmaxf(acc[mt][pt][v] + bcv[v], 0.f);
        s[v] += ok ? t : 0.f;
      }
    }
#pragma unroll
    for (int off = 1; off <= 8; off <<= 1)
#pragma unroll
      for (int v = 0; v < 8; ++v) s[v] += __shfl_xor(s[v], off, 32);
    if (lo == 0) {
#pragma unroll
      for (int v = 0; v < 8; ++v)
        pooled[bn * HC + obase + v] = (_Float16)(s[v] * (1.f / 49.f));
    }
  }
}

// ---------------------------------------------------------------- kernel 2
// One wave per 16 vertices: h1 = relu(P W1^T + b1); h2 = relu(h1 W2^T + b2);
// then 4 scalar projections per row.
__global__ __launch_bounds__(32) void k_mlp(
    const _Float16* __restrict__ pooled, const _Float16* __restrict__ W1h,
    const float* __restrict__ b1, const _Float16* __restrict__ W2h,
    const float* __restrict__ b2, const float* __restrict__ Wa,
    const float* __restrict__ ba, const float* __restrict__ Wr,
    const float* __restrict__ br, const float* __restrict__ Ww,
    const float* __restrict__ bw, float* __restrict__ out) {
  __shared__ __align__(16) _Float16 h1[16 * 256];

  const int lane  = threadIdx.x & 31;
  const int lo    = lane & 15;
  const int hiofs = (lane & 16) ? 8 : 0;
  const int hi8   = (lane & 16) ? 8 : 0;
  const int rowbase = blockIdx.x * 16;
  const v8f vz = {0.f, 0.f, 0.f, 0.f, 0.f, 0.f, 0.f, 0.f};

  // A fragments of pooled tile (M=16 rows, K=128)
  v16h af[4];
#pragma unroll
  for (int k = 0; k < 4; ++k)
    af[k] = ldfrag(pooled + (rowbase + lo) * HC + k * 32 + hiofs);

  // GEMM1: N=256 -> 16 tiles, K=128 -> 4 steps
#pragma unroll
  for (int nt = 0; nt < 16; ++nt) {
    v8f acc = vz;
#pragma unroll
    for (int k = 0; k < 4; ++k) {
      v16h bf = ldfrag(W1h + (nt * 16 + lo) * HC + k * 32 + hiofs);
      acc = wmma_f16(af[k], bf, acc);
    }
    const int j = nt * 16 + lo;
    const float bj = b1[j];
#pragma unroll
    for (int v = 0; v < 8; ++v)
      h1[(v + hi8) * 256 + j] = (_Float16)fmaxf(acc[v] + bj, 0.f);
  }
  __syncthreads();

  // GEMM2: N=128 -> 8 tiles, K=256 -> 8 steps
  v8f acc2[8];
#pragma unroll
  for (int nt = 0; nt < 8; ++nt) acc2[nt] = vz;
#pragma unroll
  for (int k = 0; k < 8; ++k) {
    v16h a2 = ldfrag(&h1[lo * 256 + k * 32 + hiofs]);
#pragma unroll
    for (int nt = 0; nt < 8; ++nt) {
      v16h bf = ldfrag(W2h + (nt * 16 + lo) * 256 + k * 32 + hiofs);
      acc2[nt] = wmma_f16(a2, bf, acc2[nt]);
    }
  }

  // bias + relu + 4 projections (fp32 VALU), cross-lane reduce per row
  float s0[8], s1[8], s2[8], s3[8];
#pragma unroll
  for (int v = 0; v < 8; ++v) { s0[v] = s1[v] = s2[v] = s3[v] = 0.f; }
#pragma unroll
  for (int nt = 0; nt < 8; ++nt) {
    const int o = nt * 16 + lo;
    const float b2o = b2[o];
    const float wa0 = Wa[o], wa1 = Wa[HC + o], wr = Wr[o], ww = Ww[o];
#pragma unroll
    for (int v = 0; v < 8; ++v) {
      float t = fmaxf(acc2[nt][v] + b2o, 0.f);
      s0[v] += t * wa0; s1[v] += t * wa1; s2[v] += t * wr; s3[v] += t * ww;
    }
  }
#pragma unroll
  for (int off = 1; off <= 8; off <<= 1) {
#pragma unroll
    for (int v = 0; v < 8; ++v) {
      s0[v] += __shfl_xor(s0[v], off, 32);
      s1[v] += __shfl_xor(s1[v], off, 32);
      s2[v] += __shfl_xor(s2[v], off, 32);
      s3[v] += __shfl_xor(s3[v], off, 32);
    }
  }
  if (lo == 0) {
    const float ba0 = ba[0], ba1 = ba[1], br0 = br[0], bw0 = bw[0];
#pragma unroll
    for (int v = 0; v < 8; ++v) {
      const int r = rowbase + hi8 + v;
      out[r * 2 + 0]  = s0[v] + ba0;        // d_angle[...,0]
      out[r * 2 + 1]  = s1[v] + ba1;        // d_angle[...,1]
      out[8192 + r]   = s2[v] + br0;        // d_radius
      out[12288 + r]  = s3[v] + bw0;        // d_width
    }
  }
}

extern "C" void kernel_launch(void* const* d_in, const int* in_sizes, int n_in,
                              void* d_out, int out_size, void* d_ws,
                              size_t ws_size, hipStream_t stream) {
  const float* feat  = (const float*)d_in[0];
  const float* verts = (const float*)d_in[1];
  const float* Wc    = (const float*)d_in[2];
  const float* bc    = (const float*)d_in[3];
  const float* W1    = (const float*)d_in[4];
  const float* b1    = (const float*)d_in[5];
  const float* W2    = (const float*)d_in[6];
  const float* b2    = (const float*)d_in[7];
  const float* Wa    = (const float*)d_in[8];
  const float* ba    = (const float*)d_in[9];
  const float* Wr    = (const float*)d_in[10];
  const float* br    = (const float*)d_in[11];
  const float* Ww    = (const float*)d_in[12];
  const float* bw    = (const float*)d_in[13];

  _Float16* wsh    = (_Float16*)d_ws;
  _Float16* Wch    = wsh;            // 32768 halves
  _Float16* W1h    = wsh + 32768;    // 32768 halves
  _Float16* W2h    = wsh + 65536;    // 32768 halves
  _Float16* pooled = wsh + 98304;    // 4096*128 halves (16B aligned)
  float* out = (float*)d_out;

  k_convert<<<dim3(384), dim3(256), 0, stream>>>(Wc, W1, W2, wsh);
  k_roi_comp<<<dim3(4096), dim3(128), 0, stream>>>(feat, verts, Wch, bc, pooled);
  k_mlp<<<dim3(256), dim3(32), 0, stream>>>(pooled, W1h, b1, W2h, b2, Wa, ba,
                                            Wr, br, Ww, bw, out);
}
